// RelativeBiasMultiHeadAttention_16269336117670
// MI455X (gfx1250) — matrix operand order repaired and checked by reading, not compiled, not run on hardware
//
#include <hip/hip_runtime.h>
#include <hip/hip_bf16.h>

#define D_MODEL   1024
#define NUM_HEADS 16
#define D_K       64
#define MAX_DIST  128
#define BATCH     4
#define SEQ       2048

typedef __attribute__((ext_vector_type(16))) _Float16 v16h;
typedef __attribute__((ext_vector_type(8)))  float    v8f;
typedef __attribute__((ext_vector_type(4)))  float    v4f;   // clang ext-vector (nontemporal-ok)

__device__ __forceinline__ v8f wmma16x16x32(v16h a, v16h b, v8f c) {
    // emits v_wmma_f32_16x16x32_f16 (wave32, fp32 accumulate)
    return __builtin_amdgcn_wmma_f32_16x16x32_f16(
        /*neg_a=*/false, a, /*neg_b=*/false, b,
        /*c_mod=*/(short)0, c, /*reuse_a=*/false, /*reuse_b=*/false);
}

// CDNA5 wave32 16-bit fragment k order used here (same for A and B): per-lane data is
// contiguous runs of row-major memory:
//   A 16x32: run  [half*16, +16)                     (one 32B run)
//   B 32x16: run  [half*16, +16)                     (one 32B run)
__device__ __forceinline__ v16h cvt16(float4 t0, float4 t1, float4 t2, float4 t3) {
    v16h r;
    r[0]  = (_Float16)t0.x; r[1]  = (_Float16)t0.y; r[2]  = (_Float16)t0.z; r[3]  = (_Float16)t0.w;
    r[4]  = (_Float16)t1.x; r[5]  = (_Float16)t1.y; r[6]  = (_Float16)t1.z; r[7]  = (_Float16)t1.w;
    r[8]  = (_Float16)t2.x; r[9]  = (_Float16)t2.y; r[10] = (_Float16)t2.z; r[11] = (_Float16)t2.w;
    r[12] = (_Float16)t3.x; r[13] = (_Float16)t3.y; r[14] = (_Float16)t3.z; r[15] = (_Float16)t3.w;
    return r;
}
__device__ __forceinline__ v16h cvt16(v4f t0, v4f t1, v4f t2, v4f t3) {
    v16h r;
    r[0]  = (_Float16)t0.x; r[1]  = (_Float16)t0.y; r[2]  = (_Float16)t0.z; r[3]  = (_Float16)t0.w;
    r[4]  = (_Float16)t1.x; r[5]  = (_Float16)t1.y; r[6]  = (_Float16)t1.z; r[7]  = (_Float16)t1.w;
    r[8]  = (_Float16)t2.x; r[9]  = (_Float16)t2.y; r[10] = (_Float16)t2.z; r[11] = (_Float16)t2.w;
    r[12] = (_Float16)t3.x; r[13] = (_Float16)t3.y; r[14] = (_Float16)t3.z; r[15] = (_Float16)t3.w;
    return r;
}

// ---------------------------------------------------------------------------
// Kernel 1: C_f16[M,N] = A_f32[M,K] @ W_f32[N,K]^T   (nn.Linear projection)
// One wave = 16x64 of C (1x4 tiles). Phase 1 issues ALL 20 b128 loads of the
// K-step; phase 2 converts and runs 4 WMMAs. No per-load waits.
// ---------------------------------------------------------------------------
__global__ void k_proj(const float* __restrict__ A, const float* __restrict__ W,
                       _Float16* __restrict__ C, int M, int N, int K) {
    const int wave = threadIdx.x >> 5;
    const int lane = threadIdx.x & 31;
    const int half = lane >> 4;
    const int lr   = lane & 15;
    const int groupsN = N >> 6;                               // col-groups of 64
    const int job = blockIdx.x * (blockDim.x >> 5) + wave;
    const int mt = job / groupsN, ng = job % groupsN;
    const size_t arow = (size_t)(mt * 16 + lr) * K;
    size_t brow[4];
#pragma unroll
    for (int j = 0; j < 4; ++j)
        brow[j] = (size_t)(ng * 64 + j * 16 + lr) * K;
    v8f acc[4] = {};
    for (int k0 = 0; k0 < K; k0 += 32) {
        __builtin_prefetch(A + arow + k0 + 128, 0, 1);        // global_prefetch_b8
        // ---- phase 1: issue all loads ----
        const float* ap = A + arow + k0 + half * 16;
        float4 ta0 = *(const float4*)(ap);
        float4 ta1 = *(const float4*)(ap + 4);
        float4 ta2 = *(const float4*)(ap + 8);
        float4 ta3 = *(const float4*)(ap + 12);
        float4 tb[4][4];
#pragma unroll
        for (int j = 0; j < 4; ++j) {
            const float* bp = W + brow[j] + k0 + half * 16;
#pragma unroll
            for (int u = 0; u < 4; ++u) tb[j][u] = *(const float4*)(bp + 4 * u);
        }
        // ---- phase 2: convert + WMMA ----
        v16h a = cvt16(ta0, ta1, ta2, ta3);
#pragma unroll
        for (int j = 0; j < 4; ++j) {
            v16h b = cvt16(tb[j][0], tb[j][1], tb[j][2], tb[j][3]);
            acc[j] = wmma16x16x32(a, b, acc[j]);
        }
    }
#pragma unroll
    for (int j = 0; j < 4; ++j) {
        const int c = ng * 64 + j * 16 + lr;
#pragma unroll
        for (int v = 0; v < 8; ++v) {
            const int r = mt * 16 + v + 8 * half;             // C/D layout row
            C[(size_t)r * N + c] = (_Float16)acc[j][v];
        }
    }
}

// ---------------------------------------------------------------------------
// Kernel 2: scores[b,h,q,k] = (Q_h . K_h)/8 + bias[rel] ; mask==0 -> -1e9
// grid.y = b*H+h ; one wave = 16x64 region. Fragments are pure b128 copies,
// all 10 loads of a K-step issued before any WMMA.
// ---------------------------------------------------------------------------
__global__ void k_scores(const _Float16* __restrict__ Q, const _Float16* __restrict__ Kp,
                         const int* __restrict__ mask, const float* __restrict__ bias_table,
                         float* __restrict__ attn) {
    const int wave = threadIdx.x >> 5;
    const int lane = threadIdx.x & 31;
    const int half = lane >> 4;
    const int lr   = lane & 15;
    const int bh = blockIdx.y;
    const int b  = bh >> 4;
    const int h  = bh & 15;
    const int groupsN = SEQ >> 6;                             // 32 col-groups of 64
    const int job = blockIdx.x * (blockDim.x >> 5) + wave;
    const int mt = job / groupsN, ng = job % groupsN;
    const size_t headoff = (size_t)h * D_K;
    const size_t abase = (size_t)(b * SEQ + mt * 16 + lr) * D_MODEL + headoff;
    size_t bbase[4];
#pragma unroll
    for (int j = 0; j < 4; ++j)
        bbase[j] = (size_t)(b * SEQ + ng * 64 + j * 16 + lr) * D_MODEL + headoff;
    v8f acc[4] = {};
#pragma unroll
    for (int k0 = 0; k0 < D_K; k0 += 32) {
        // ---- phase 1: issue all loads ----
        const _Float16* ap = Q + abase + k0 + half * 16;
        uint4 ta0 = *(const uint4*)(ap);
        uint4 ta1 = *(const uint4*)(ap + 8);
        uint4 tb[4][2];
#pragma unroll
        for (int j = 0; j < 4; ++j) {
            const _Float16* bp = Kp + bbase[j] + k0 + half * 16;
            tb[j][0] = *(const uint4*)(bp);
            tb[j][1] = *(const uint4*)(bp + 8);
        }
        // ---- phase 2: WMMA ----
        v16h a;
        ((uint4*)&a)[0] = ta0;
        ((uint4*)&a)[1] = ta1;
#pragma unroll
        for (int j = 0; j < 4; ++j) {
            v16h bf;
            ((uint4*)&bf)[0] = tb[j][0];
            ((uint4*)&bf)[1] = tb[j][1];
            acc[j] = wmma16x16x32(a, bf, acc[j]);
        }
    }
    float* out = attn + (size_t)bh * SEQ * SEQ;
#pragma unroll
    for (int j = 0; j < 4; ++j) {
        const int kcol = ng * 64 + j * 16 + lr;
#pragma unroll
        for (int v = 0; v < 8; ++v) {
            const int q = mt * 16 + v + 8 * half;
            int rel = q - kcol + (MAX_DIST - 1);
            rel = rel < 0 ? 0 : (rel > 2 * MAX_DIST - 2 ? 2 * MAX_DIST - 2 : rel);
            float s = acc[j][v] * 0.125f + bias_table[rel * NUM_HEADS + h];
            if (mask[(size_t)q * SEQ + kcol] == 0) s = -1e9f;
            out[(size_t)q * SEQ + kcol] = s;
        }
    }
}

// ---------------------------------------------------------------------------
// Kernel 3: row softmax, in place. One block per row; row staged in LDS so
// attn is read exactly once and written exactly once (bandwidth floor).
// ---------------------------------------------------------------------------
__global__ void k_softmax(float* __restrict__ attn) {
    __shared__ float buf[SEQ];
    __shared__ float red[256];
    float* p = attn + (size_t)blockIdx.x * SEQ;
    const int t = threadIdx.x;
    float m = -3.0e38f;
    for (int i = t; i < SEQ; i += 256) { float x = p[i]; buf[i] = x; m = fmaxf(m, x); }
    red[t] = m; __syncthreads();
    for (int s = 128; s > 0; s >>= 1) {
        if (t < s) red[t] = fmaxf(red[t], red[t + s]);
        __syncthreads();
    }
    m = red[0]; __syncthreads();
    float sum = 0.f;
    for (int i = t; i < SEQ; i += 256) { float e = __expf(buf[i] - m); buf[i] = e; sum += e; }
    red[t] = sum; __syncthreads();
    for (int s = 128; s > 0; s >>= 1) {
        if (t < s) red[t] += red[t + s];
        __syncthreads();
    }
    const float inv = 1.0f / red[0];
    for (int i = t; i < SEQ; i += 256) p[i] = buf[i] * inv;
}

// ---------------------------------------------------------------------------
// Kernel 4: ctx[b,q,h*64+d] = sum_k attn[b,h,q,k] * V[b,k,h*64+d]
// One wave = 16 q-rows x full d_k=64. attn rows are streamed (non-temporal,
// each element read exactly once); all V loads of a K-step issued up front.
// ---------------------------------------------------------------------------
__global__ void k_attnV(const float* __restrict__ attn, const _Float16* __restrict__ V,
                        _Float16* __restrict__ ctx) {
    const int wave = threadIdx.x >> 5;
    const int lane = threadIdx.x & 31;
    const int half = lane >> 4;
    const int lr   = lane & 15;
    const int bh = blockIdx.y;
    const int b  = bh >> 4;
    const int h  = bh & 15;
    const int mt = blockIdx.x * (blockDim.x >> 5) + wave;     // 16-row q tile
    const float* A = attn + (size_t)bh * SEQ * SEQ;
    const size_t arow = (size_t)(mt * 16 + lr) * SEQ;
    const size_t vbase = (size_t)h * D_K + lr;
    v8f acc[4] = {};
    for (int k0 = 0; k0 < SEQ; k0 += 32) {
        __builtin_prefetch(A + arow + k0 + 128, 0, 1);        // global_prefetch_b8
        // ---- phase 1: issue all loads ----
        const float* ap = A + arow + k0 + half * 16;
        v4f ta0 = __builtin_nontemporal_load((const v4f*)(ap));
        v4f ta1 = __builtin_nontemporal_load((const v4f*)(ap + 4));
        v4f ta2 = __builtin_nontemporal_load((const v4f*)(ap + 8));
        v4f ta3 = __builtin_nontemporal_load((const v4f*)(ap + 12));
        v16h tb[4];
#pragma unroll
        for (int j = 0; j < 4; ++j) {
#pragma unroll
            for (int v = 0; v < 8; ++v) {
                const int kb = k0 + half * 16 + 2 * v;        // B layout K pair
                tb[j][2 * v]     = V[(size_t)(b * SEQ + kb)     * D_MODEL + vbase + j * 16];
                tb[j][2 * v + 1] = V[(size_t)(b * SEQ + kb + 1) * D_MODEL + vbase + j * 16];
            }
        }
        // ---- phase 2: convert + WMMA ----
        v16h a = cvt16(ta0, ta1, ta2, ta3);
#pragma unroll
        for (int j = 0; j < 4; ++j)
            acc[j] = wmma16x16x32(a, tb[j], acc[j]);
    }
#pragma unroll
    for (int j = 0; j < 4; ++j) {
#pragma unroll
        for (int v = 0; v < 8; ++v) {
            const int q = mt * 16 + v + 8 * half;
            ctx[(size_t)(b * SEQ + q) * D_MODEL + (size_t)h * D_K + j * 16 + lr] =
                (_Float16)acc[j][v];
        }
    }
}

// ---------------------------------------------------------------------------
// Kernel 5: out_f32[M,N] = ctx_f16[M,K] @ Wo_f32[N,K]^T + bo[N]
// Same two-phase 1x4 tiling as k_proj; fp32 output with bias.
// ---------------------------------------------------------------------------
__global__ void k_out(const _Float16* __restrict__ A, const float* __restrict__ W,
                      const float* __restrict__ bo, float* __restrict__ C,
                      int M, int N, int K) {
    const int wave = threadIdx.x >> 5;
    const int lane = threadIdx.x & 31;
    const int half = lane >> 4;
    const int lr   = lane & 15;
    const int groupsN = N >> 6;
    const int job = blockIdx.x * (blockDim.x >> 5) + wave;
    const int mt = job / groupsN, ng = job % groupsN;
    const size_t arow = (size_t)(mt * 16 + lr) * K;
    size_t brow[4];
#pragma unroll
    for (int j = 0; j < 4; ++j)
        brow[j] = (size_t)(ng * 64 + j * 16 + lr) * K;
    v8f acc[4] = {};
    for (int k0 = 0; k0 < K; k0 += 32) {
        __builtin_prefetch(A + arow + k0 + 256, 0, 1);        // global_prefetch_b8
        // ---- phase 1: issue all loads ----
        const _Float16* ap = A + arow + k0 + half * 16;
        uint4 ta0 = *(const uint4*)(ap);
        uint4 ta1 = *(const uint4*)(ap + 8);
        float4 tb[4][4];
#pragma unroll
        for (int j = 0; j < 4; ++j) {
            const float* bp = W + brow[j] + k0 + half * 16;
#pragma unroll
            for (int u = 0; u < 4; ++u) tb[j][u] = *(const float4*)(bp + 4 * u);
        }
        // ---- phase 2: convert + WMMA ----
        v16h a;
        ((uint4*)&a)[0] = ta0;
        ((uint4*)&a)[1] = ta1;
#pragma unroll
        for (int j = 0; j < 4; ++j) {
            v16h b = cvt16(tb[j][0], tb[j][1], tb[j][2], tb[j][3]);
            acc[j] = wmma16x16x32(a, b, acc[j]);
        }
    }
#pragma unroll
    for (int j = 0; j < 4; ++j) {
        const int c = ng * 64 + j * 16 + lr;
        const float bias = bo[c];
#pragma unroll
        for (int v = 0; v < 8; ++v) {
            const int r = mt * 16 + v + 8 * half;
            C[(size_t)r * N + c] = acc[j][v] + bias;
        }
    }
}

extern "C" void kernel_launch(void* const* d_in, const int* in_sizes, int n_in,
                              void* d_out, int out_size, void* d_ws, size_t ws_size,
                              hipStream_t stream) {
    const float* query = (const float*)d_in[0];
    const float* key   = (const float*)d_in[1];
    const float* value = (const float*)d_in[2];
    const int*   mask  = (const int*)d_in[3];
    const float* Wq = (const float*)d_in[4];
    const float* Wk = (const float*)d_in[5];
    const float* Wv = (const float*)d_in[6];
    const float* Wo = (const float*)d_in[7];
    const float* bo = (const float*)d_in[8];
    const float* bias_table = (const float*)d_in[9];

    float* out  = (float*)d_out;
    float* attn = out + (size_t)BATCH * SEQ * D_MODEL;   // outputs concatenated

    const size_t n_proj = (size_t)BATCH * SEQ * D_MODEL; // 8.4M elems
    _Float16* Qbuf = (_Float16*)d_ws;                    // 16 MB
    _Float16* Kbuf = Qbuf + n_proj;                      // 16 MB
    _Float16* Vbuf = Kbuf + n_proj;                      // 16 MB
    _Float16* Ctx  = Qbuf;   // Q is dead after k_scores; reuse (stream-ordered)

    const int M = BATCH * SEQ;                           // 8192
    const dim3 blk(128);                                 // 4 waves / block
    const int projBlocks = (M / 16) * (D_MODEL / 64) / 4; // 2048

    k_proj<<<projBlocks, blk, 0, stream>>>(query, Wq, Qbuf, M, D_MODEL, D_MODEL);
    k_proj<<<projBlocks, blk, 0, stream>>>(key,   Wk, Kbuf, M, D_MODEL, D_MODEL);
    k_proj<<<projBlocks, blk, 0, stream>>>(value, Wv, Vbuf, M, D_MODEL, D_MODEL);

    dim3 sgrid((SEQ / 16) * (SEQ / 64) / 4, BATCH * NUM_HEADS);   // 1024 x 64
    k_scores<<<sgrid, blk, 0, stream>>>(Qbuf, Kbuf, mask, bias_table, attn);

    k_softmax<<<BATCH * NUM_HEADS * SEQ, 256, 0, stream>>>(attn);

    dim3 vgrid((SEQ / 16) / 4, BATCH * NUM_HEADS);                // 32 x 64
    k_attnV<<<vgrid, blk, 0, stream>>>(attn, Vbuf, Ctx);

    k_out<<<projBlocks, blk, 0, stream>>>(Ctx, Wo, bo, out, M, D_MODEL, D_MODEL);
}
